// JointPooling_Node_68702296866864
// MI455X (gfx1250) — matrix operand order repaired and checked by reading, not compiled, not run on hardware
//
#include <hip/hip_runtime.h>
#include <math.h>

typedef __attribute__((ext_vector_type(16))) _Float16 v16h;
typedef __attribute__((ext_vector_type(8)))  _Float16 v8h;
typedef __attribute__((ext_vector_type(8)))  float    v8f;

#define Gn   16
#define Nn   512
#define En   2048
#define CH   64
#define KEEPN 256
#define PITERS 64
#define TILE 128
#define KCH  128

// ---------------- elementwise / scatter helpers ----------------

__global__ void zero_f32(float* p, int n) {
    int t = blockIdx.x * blockDim.x + threadIdx.x;
    if (t < n) p[t] = 0.f;
}

// L(z)[dst,c] += w[e] * z[src,c]   (64 channels per edge)
__global__ void scatter_L(const float* __restrict__ z, const int* __restrict__ src,
                          const int* __restrict__ dst, const float* __restrict__ w,
                          float* __restrict__ out, int nEdges) {
    int t = blockIdx.x * blockDim.x + threadIdx.x;
    int e = t >> 6, c = t & 63;
    if (e >= nEdges) return;
    atomicAdd(&out[(size_t)dst[e] * CH + c], w[e] * z[(size_t)src[e] * CH + c]);
}

__global__ void tx1_k(const float* __restrict__ x, const float* __restrict__ Lx,
                      float* __restrict__ T, int n) {
    int t = blockIdx.x * blockDim.x + threadIdx.x;
    if (t < n) T[t] = x[t] - Lx[t];
}

// score[i] = sigmoid( sum_c x*W0 + T1*W1 + ((-LT1 + 3*T1 - x)/2)*W2 + b )
__global__ void score_k(const float* __restrict__ x, const float* __restrict__ T,
                        const float* __restrict__ L, const float* __restrict__ W,
                        const float* __restrict__ b, float* __restrict__ score, int nRows) {
    int i = blockIdx.x * blockDim.x + threadIdx.x;
    if (i >= nRows) return;
    float acc = 0.f;
    #pragma unroll 8
    for (int c = 0; c < CH; ++c) {
        float xv = x[(size_t)i * CH + c];
        float tv = T[(size_t)i * CH + c];
        float lv = L[(size_t)i * CH + c];
        float t2 = 0.5f * (-lv + 3.f * tv - xv);
        acc += xv * W[c] + tv * W[CH + c] + t2 * W[2 * CH + c];
    }
    acc += b[0];
    score[i] = 1.f / (1.f + expf(-acc));
}

__global__ void aug_init_k(const float* __restrict__ sn, float* __restrict__ aug, int n) {
    int t = blockIdx.x * blockDim.x + threadIdx.x;
    if (t < n) aug[t] = sn[t];
}

// aug[g, u[e]] += se[e] ; aug[g, v[e]] += se[e]
__global__ void aug_scatter_k(const float* __restrict__ se, const int* __restrict__ eu,
                              const int* __restrict__ ev, float* __restrict__ aug) {
    int t = blockIdx.x * blockDim.x + threadIdx.x;
    if (t >= Gn * En) return;
    int base = (t >> 11) * Nn;
    float s = se[t];
    atomicAdd(&aug[base + eu[t]], s);
    atomicAdd(&aug[base + ev[t]], s);
}

// top-KEEP selection via stable rank (matches argsort(-aug) tie rule: lower index first)
__global__ __launch_bounds__(Nn) void mask_k(const float* __restrict__ aug, int* __restrict__ mask) {
    int g = blockIdx.x, tid = threadIdx.x;
    __shared__ float a[Nn];
    a[tid] = aug[g * Nn + tid];
    __syncthreads();
    float mine = a[tid];
    int rank = 0;
    for (int j = 0; j < Nn; ++j) {
        float aj = a[j];
        rank += (aj > mine) || (aj == mine && j < tid);
    }
    mask[g * Nn + tid] = (rank < KEEPN) ? 1 : 0;
}

__global__ void keep_k(const int* __restrict__ eu, const int* __restrict__ ev,
                       const int* __restrict__ mask, int* __restrict__ keep) {
    int t = blockIdx.x * blockDim.x + threadIdx.x;
    if (t >= Gn * En) return;
    int base = (t >> 11) * Nn;
    keep[t] = mask[base + eu[t]] & mask[base + ev[t]];
}

__global__ void xsc_out_k(const float* __restrict__ x, const float* __restrict__ score,
                          const int* __restrict__ m, float* __restrict__ out, int n) {
    int t = blockIdx.x * blockDim.x + threadIdx.x;
    if (t >= n) return;
    int i = t >> 6;
    out[t] = x[t] * score[i] * (m[i] ? 1.f : 0.f);
}

// ---------------- power iteration on L0 = Bm Bm^T via sparse LDS ops ----------------

__global__ __launch_bounds__(Nn) void power_k(const int* __restrict__ eu, const int* __restrict__ ev,
                                              const int* __restrict__ keep, float* __restrict__ scale) {
    int g = blockIdx.x, tid = threadIdx.x;
    __shared__ float w[Nn];
    __shared__ float y[Nn];
    __shared__ float red[17];
    const int gE = g * En;
    int u0[4], v0[4]; bool kp[4];
    #pragma unroll
    for (int k = 0; k < 4; ++k) {
        int e = tid + Nn * k;
        u0[k] = eu[gE + e]; v0[k] = ev[gE + e]; kp[k] = keep[gE + e] != 0;
    }
    w[tid] = 0.044194173824159216f; // 1/sqrt(512)
    __syncthreads();
    for (int it = 0; it < PITERS; ++it) {
        y[tid] = 0.f;
        __syncthreads();
        #pragma unroll
        for (int k = 0; k < 4; ++k)
            if (kp[k]) {
                float ye = w[v0[k]] - w[u0[k]];         // (Bm^T w)[e]
                atomicAdd(&y[u0[k]], -ye);              // Bm * ye
                atomicAdd(&y[v0[k]],  ye);
            }
        __syncthreads();
        float val = y[tid];
        float s = val * val;
        for (int o = 16; o > 0; o >>= 1) s += __shfl_down(s, o, 32);
        if ((tid & 31) == 0) red[tid >> 5] = s;
        __syncthreads();
        if (tid < 32) {
            float r = (tid < 16) ? red[tid] : 0.f;
            for (int o = 8; o > 0; o >>= 1) r += __shfl_down(r, o, 32);
            if (tid == 0) red[16] = 1.f / (sqrtf(r) + 1e-12f);
        }
        __syncthreads();
        w[tid] = val * red[16];
        __syncthreads();
    }
    // Rayleigh quotient lam = w . (L0 w)
    y[tid] = 0.f;
    __syncthreads();
    #pragma unroll
    for (int k = 0; k < 4; ++k)
        if (kp[k]) {
            float ye = w[v0[k]] - w[u0[k]];
            atomicAdd(&y[u0[k]], -ye);
            atomicAdd(&y[v0[k]],  ye);
        }
    __syncthreads();
    float s = w[tid] * y[tid];
    for (int o = 16; o > 0; o >>= 1) s += __shfl_down(s, o, 32);
    if ((tid & 31) == 0) red[tid >> 5] = s;
    __syncthreads();
    if (tid == 0) {
        float lam = 0.f;
        for (int i = 0; i < 16; ++i) lam += red[i];
        scale[g] = 2.f / (lam + 1e-12f);
    }
}

// ---------------- WMMA GEMMs: L0 = Bm Bm^T, L1 = Bm^T Bm ----------------
// Tiles of Bm are built in LDS straight from the edge list (<=2 nonzeros per column).

__device__ __forceinline__ void wmma_tile_accum(const _Float16* __restrict__ A,
                                                const _Float16* __restrict__ B,
                                                int wv, int lane, v8f acc[8]) {
    const int m    = lane & 15;
    const int offA = (lane < 16) ? 0 : 16;  // A 16x32 f16 per-lane K pattern (contiguous k = 16h + s)
    const int offB = (lane < 16) ? 0 : 16;  // B 32x16 f16 per-lane K pattern
    #pragma unroll
    for (int kk = 0; kk < KCH; kk += 32) {
        const _Float16* ra = A + (wv * 16 + m) * KCH + kk;
        v8h a0 = *(const v8h*)(ra + offA);
        v8h a1 = *(const v8h*)(ra + offA + 8);
        v16h av = __builtin_shufflevector(a0, a1, 0,1,2,3,4,5,6,7,8,9,10,11,12,13,14,15);
        #pragma unroll
        for (int t = 0; t < 8; ++t) {
            const _Float16* rb = B + (t * 16 + m) * KCH + kk + offB;
            v16h bv = *(const v16h*)rb;
            acc[t] = __builtin_amdgcn_wmma_f32_16x16x32_f16(false, av, false, bv,
                                                            (short)0, acc[t], false, false);
        }
    }
}

// L1[e,f] = sum_n Bm[n,e] Bm[n,f]   (rows/cols = edges, K = nodes)
__global__ __launch_bounds__(256) void l1_gemm(const int* __restrict__ eu, const int* __restrict__ ev,
                                               const int* __restrict__ keep,
                                               const float* __restrict__ scale,
                                               float* __restrict__ out) {
    const int g = blockIdx.z, tid = threadIdx.x;
    const int f0 = blockIdx.x * TILE, e0 = blockIdx.y * TILE;
    const int lane = tid & 31, wv = tid >> 5;
    const int gE = g * En;
    __shared__ _Float16 tile[2 * TILE * KCH];       // 64 KB
    _Float16* A = tile;                             // [row e][k n]
    _Float16* B = tile + TILE * KCH;                // [col f][k n] (column-major)
    v8f acc[8];
    #pragma unroll
    for (int t = 0; t < 8; ++t) acc[t] = (v8f){0,0,0,0,0,0,0,0};

    for (int k0 = 0; k0 < Nn; k0 += KCH) {
        unsigned int* z = (unsigned int*)tile;
        for (int i = tid; i < TILE * KCH; i += 256) z[i] = 0u;
        __syncthreads();
        if (tid < TILE) {                            // build A row (edge e0+tid)
            int e = e0 + tid;
            if (keep[gE + e]) {
                int un = eu[gE + e] - k0, vn = ev[gE + e] - k0;
                if ((unsigned)un < KCH) A[tid * KCH + un] = (_Float16)(-1.0f);
                if ((unsigned)vn < KCH) A[tid * KCH + vn] = (_Float16)( 1.0f);
            }
        } else {                                     // build B column (edge f0+tid-128)
            int r = tid - TILE, f = f0 + r;
            if (keep[gE + f]) {
                int un = eu[gE + f] - k0, vn = ev[gE + f] - k0;
                if ((unsigned)un < KCH) B[r * KCH + un] = (_Float16)(-1.0f);
                if ((unsigned)vn < KCH) B[r * KCH + vn] = (_Float16)( 1.0f);
            }
        }
        __syncthreads();
        wmma_tile_accum(A, B, wv, lane, acc);
        __syncthreads();
    }
    const float sc = scale[g];
    const int row0 = e0 + wv * 16 + ((lane < 16) ? 0 : 8);
    float* og = out + (size_t)g * En * En;
    #pragma unroll
    for (int t = 0; t < 8; ++t) {
        int c = f0 + t * 16 + (lane & 15);
        #pragma unroll
        for (int r = 0; r < 8; ++r)
            og[(size_t)(row0 + r) * En + c] = acc[t][r] * sc;
    }
}

// L0[i,j] = sum_e Bm[i,e] Bm[j,e]   (rows/cols = nodes, K = edges)
__global__ __launch_bounds__(256) void l0_gemm(const int* __restrict__ eu, const int* __restrict__ ev,
                                               const int* __restrict__ keep,
                                               const float* __restrict__ scale,
                                               float* __restrict__ out) {
    const int g = blockIdx.z, tid = threadIdx.x;
    const int j0 = blockIdx.x * TILE, n0 = blockIdx.y * TILE;
    const int lane = tid & 31, wv = tid >> 5;
    const int gE = g * En;
    __shared__ _Float16 tile[2 * TILE * KCH];
    _Float16* A = tile;                              // [row n][k e]
    _Float16* B = tile + TILE * KCH;                 // [col j][k e]
    v8f acc[8];
    #pragma unroll
    for (int t = 0; t < 8; ++t) acc[t] = (v8f){0,0,0,0,0,0,0,0};

    for (int k0 = 0; k0 < En; k0 += KCH) {
        unsigned int* z = (unsigned int*)tile;
        for (int i = tid; i < TILE * KCH; i += 256) z[i] = 0u;
        __syncthreads();
        if (tid < KCH) {                             // one thread per K-edge (owns column tid)
            int e = k0 + tid;
            if (keep[gE + e]) {
                int un = eu[gE + e], vn = ev[gE + e];
                int ra = un - n0; if ((unsigned)ra < TILE) A[ra * KCH + tid] = (_Float16)(-1.0f);
                int rv = vn - n0; if ((unsigned)rv < TILE) A[rv * KCH + tid] = (_Float16)( 1.0f);
                int ca = un - j0; if ((unsigned)ca < TILE) B[ca * KCH + tid] = (_Float16)(-1.0f);
                int cv = vn - j0; if ((unsigned)cv < TILE) B[cv * KCH + tid] = (_Float16)( 1.0f);
            }
        }
        __syncthreads();
        wmma_tile_accum(A, B, wv, lane, acc);
        __syncthreads();
    }
    const float sc = scale[g];
    const int row0 = n0 + wv * 16 + ((lane < 16) ? 0 : 8);
    float* og = out + (size_t)g * Nn * Nn;
    #pragma unroll
    for (int t = 0; t < 8; ++t) {
        int c = j0 + t * 16 + (lane & 15);
        #pragma unroll
        for (int r = 0; r < 8; ++r)
            og[(size_t)(row0 + r) * Nn + c] = acc[t][r] * sc;
    }
}

// ---------------- launch ----------------

extern "C" void kernel_launch(void* const* d_in, const int* in_sizes, int n_in,
                              void* d_out, int out_size, void* d_ws, size_t ws_size,
                              hipStream_t stream) {
    const float* x_n  = (const float*)d_in[0];   // 8192 x 64
    const float* x_e  = (const float*)d_in[1];   // 32768 x 64
    const int*   eu   = (const int*)  d_in[2];   // (16,2048)
    const int*   ev   = (const int*)  d_in[3];
    const int*   srcn = (const int*)  d_in[4];   // 65536
    const int*   dstn = (const int*)  d_in[5];
    const float* ewn  = (const float*)d_in[6];
    const int*   srce = (const int*)  d_in[7];   // 131072
    const int*   dste = (const int*)  d_in[8];
    const float* ewe  = (const float*)d_in[9];
    const float* W_n  = (const float*)d_in[10];  // (3,64)
    const float* b_n  = (const float*)d_in[11];
    const float* W_e  = (const float*)d_in[12];
    const float* b_e  = (const float*)d_in[13];

    // workspace layout (floats)
    float* ws      = (float*)d_ws;
    float* L_buf   = ws;                         // 2097152
    float* T_buf   = ws + 2097152;               // 2097152
    float* score_n = ws + 4194304;               // 8192
    float* score_e = ws + 4202496;               // 32768
    float* aug     = ws + 4235264;               // 8192
    float* scale   = ws + 4243456;               // 16
    int*   mask_n  = (int*)(ws + 4243472);       // 8192
    int*   keep_e  = (int*)(ws + 4251664);       // 32768

    // output layout (concatenated tuple)
    float* out_xn = (float*)d_out;               //  524288
    float* out_L0 = out_xn + 524288;             // 4194304
    float* out_xe = out_L0 + 4194304;            // 2097152
    float* out_L1 = out_xe + 2097152;            // 67108864

    const int B = 256;
    auto grid = [](int n, int b) { return (n + b - 1) / b; };

    // --- node score: HodgeLaguerre(K=3) + sigmoid ---
    zero_f32 <<<grid(524288, B), B, 0, stream>>>(L_buf, 524288);
    scatter_L<<<grid(65536 * CH, B), B, 0, stream>>>(x_n, srcn, dstn, ewn, L_buf, 65536);
    tx1_k    <<<grid(524288, B), B, 0, stream>>>(x_n, L_buf, T_buf, 524288);
    zero_f32 <<<grid(524288, B), B, 0, stream>>>(L_buf, 524288);
    scatter_L<<<grid(65536 * CH, B), B, 0, stream>>>(T_buf, srcn, dstn, ewn, L_buf, 65536);
    score_k  <<<grid(8192, B), B, 0, stream>>>(x_n, T_buf, L_buf, W_n, b_n, score_n, 8192);

    // --- edge score ---
    zero_f32 <<<grid(2097152, B), B, 0, stream>>>(L_buf, 2097152);
    scatter_L<<<grid(131072 * CH, B), B, 0, stream>>>(x_e, srce, dste, ewe, L_buf, 131072);
    tx1_k    <<<grid(2097152, B), B, 0, stream>>>(x_e, L_buf, T_buf, 2097152);
    zero_f32 <<<grid(2097152, B), B, 0, stream>>>(L_buf, 2097152);
    scatter_L<<<grid(131072 * CH, B), B, 0, stream>>>(T_buf, srce, dste, ewe, L_buf, 131072);
    score_k  <<<grid(32768, B), B, 0, stream>>>(x_e, T_buf, L_buf, W_e, b_e, score_e, 32768);

    // --- augmented node score, top-K mask, edge keep ---
    aug_init_k   <<<grid(8192, B), B, 0, stream>>>(score_n, aug, 8192);
    aug_scatter_k<<<grid(Gn * En, B), B, 0, stream>>>(score_e, eu, ev, aug);
    mask_k       <<<Gn, Nn, 0, stream>>>(aug, mask_n);
    keep_k       <<<grid(Gn * En, B), B, 0, stream>>>(eu, ev, mask_n, keep_e);

    // --- masked feature outputs ---
    xsc_out_k<<<grid(524288, B), B, 0, stream>>>(x_n, score_n, mask_n, out_xn, 524288);
    xsc_out_k<<<grid(2097152, B), B, 0, stream>>>(x_e, score_e, keep_e, out_xe, 2097152);

    // --- lambda_max(L0) via sparse power iteration, then scale = 2/lam ---
    power_k<<<Gn, Nn, 0, stream>>>(eu, ev, keep_e, scale);

    // --- scaled Laplacians via WMMA GEMM over LDS-built incidence tiles ---
    l0_gemm<<<dim3(Nn / TILE, Nn / TILE, Gn), 256, 0, stream>>>(eu, ev, keep_e, scale, out_L0);
    l1_gemm<<<dim3(En / TILE, En / TILE, Gn), 256, 0, stream>>>(eu, ev, keep_e, scale, out_L1);
}
